// RoPEMultiHeadSelfAttention_57071525429825
// MI455X (gfx1250) — matrix operand order repaired and checked by reading, not compiled, not run on hardware
//
#include <hip/hip_runtime.h>

typedef unsigned short ushort_t;
typedef __attribute__((ext_vector_type(16))) __bf16 v16bf;
typedef __attribute__((ext_vector_type(8)))  __bf16 v8bf;
typedef __attribute__((ext_vector_type(8)))  float  v8f;
typedef __attribute__((ext_vector_type(4)))  unsigned v4u;
typedef __attribute__((ext_vector_type(8)))  int      v8i;
typedef __attribute__((ext_vector_type(4)))  int      v4i;

#define D_MODEL 1024
#define SEQ     2048
#define BATCH   2
#define NHEADS  16
#define HDIM    64
#define MTOT    (BATCH * SEQ)          // 4096 rows
#define NELEM   (MTOT * D_MODEL)       // 4 M elements

static __device__ __forceinline__ ushort_t f2bfu(float f) {
    unsigned u = __builtin_bit_cast(unsigned, f);
    u += 0x7fffu + ((u >> 16) & 1u);   // round-to-nearest-even
    return (ushort_t)(u >> 16);
}

static __device__ __forceinline__ v16bf cat8(v8bf a, v8bf b) {
    return __builtin_shufflevector(a, b, 0,1,2,3,4,5,6,7,8,9,10,11,12,13,14,15);
}

static __device__ __forceinline__ v8f wmma_bf16(v16bf a, v16bf b, v8f c) {
    return __builtin_amdgcn_wmma_f32_16x16x32_bf16(false, a, false, b, (short)0, c, false, false);
}

// ---- Tensor Data Mover: 2-D bf16 tile (tile_w x tile_h elems, row stride in elems) -> LDS
// D# layout per CDNA5 ISA 8.3/8.4: group0 {count, lds_addr, global_addr, type=2},
// group1 {wg_mask=0, data_size=1(2B), tensor dims, tile dims, dim0 stride}. Groups 2/3 zero (2-D).
// This toolchain exposes the 6-arg builtin (extra int32x8 group before cpol).
static __device__ __forceinline__ void tdm_load_2d_bf16(unsigned lds_off, const ushort_t* gptr,
                                                        unsigned tile_w, unsigned tile_h,
                                                        unsigned stride_elems) {
    unsigned long long ga = (unsigned long long)(const void*)gptr;
    v4u g0;
    g0[0] = 1u;                                        // count=1, user descriptor
    g0[1] = lds_off;                                   // LDS byte address
    g0[2] = (unsigned)(ga & 0xffffffffu);              // global_addr[31:0]
    g0[3] = (unsigned)((ga >> 32) & 0x01ffffffu)       // global_addr[56:32]
          | (2u << 30);                                // type = 2 ("image")
    unsigned td0 = stride_elems;                       // tensor_dim0 (OOB bound, full row)
    unsigned td1 = 1u << 20;                           // tensor_dim1 (generous OOB bound)
    unsigned long long st0 = stride_elems;             // tensor_dim0_stride (elems)
    v8i g1;
    g1[0] = (int)(1u << 16);                           // wg_mask=0, data_size=1 (2 bytes)
    g1[1] = (int)((td0 & 0xffffu) << 16);              // tensor_dim0[15:0]
    g1[2] = (int)(((td0 >> 16) & 0xffffu) | ((td1 & 0xffffu) << 16));
    g1[3] = (int)(((td1 >> 16) & 0xffffu) | ((tile_w & 0xffffu) << 16)); // tile_dim0
    g1[4] = (int)(tile_h & 0xffffu);                   // tile_dim1; tile_dim2=0
    g1[5] = (int)(st0 & 0xffffffffu);
    g1[6] = (int)((st0 >> 32) & 0xffffu);              // dim1_stride=0 (2-D)
    g1[7] = 0;
    v4i z4 = (v4i){0, 0, 0, 0};
    v8i z8 = (v8i){0, 0, 0, 0, 0, 0, 0, 0};
    __builtin_amdgcn_tensor_load_to_lds(g0, g1, z4, z4, z8, 0);
}

static __device__ __forceinline__ unsigned lds_offset_of(const void* p) {
    // generic LDS pointer = {SHARED_BASE, offset[31:0]} -> low 32 bits are the LDS byte offset
    return (unsigned)(size_t)p;
}

// ---------------------------------------------------------------- convert f32 -> bf16
__global__ void convert_bf16(const float* __restrict__ in, ushort_t* __restrict__ out, int n) {
    for (int i = blockIdx.x * blockDim.x + threadIdx.x; i < n; i += gridDim.x * blockDim.x)
        out[i] = f2bfu(in[i]);
}

// ---------------------------------------------------------------- GEMM: C[M,N] = A[M,K] * W[N,K]^T
// B-panel (64 rows of W x 32 k) staged by TDM into LDS, double-buffered; A streamed from global.
// mode 0: f32 C row-major.  mode 2: bf16 V transposed per head: VT[(bh*64+d)*SEQ + s]
__global__ __launch_bounds__(128) void gemm_bf16_wmma(const ushort_t* __restrict__ A,
                                                      const ushort_t* __restrict__ W,
                                                      float* __restrict__ C,
                                                      ushort_t* __restrict__ VT,
                                                      int mode) {
    __shared__ __attribute__((aligned(64))) ushort_t ldsB[2][64 * 32];

    const int wave = threadIdx.x >> 5;
    const int lane = threadIdx.x & 31;
    const int half = lane >> 4;
    const int l15  = lane & 15;
    const int m0   = blockIdx.y * 64 + wave * 16;
    const int n0   = blockIdx.x * 64;
    const int NK   = D_MODEL / 32;

    v8f acc[4];
    #pragma unroll
    for (int t = 0; t < 4; ++t) acc[t] = (v8f){};

    const ushort_t* arow  = A + (size_t)(m0 + l15) * D_MODEL;
    const ushort_t* wbase = W + (size_t)n0 * D_MODEL;

    if (threadIdx.x == 0)
        tdm_load_2d_bf16(lds_offset_of(&ldsB[0][0]), wbase, 32, 64, D_MODEL);

    for (int kb = 0; kb < NK; ++kb) {
        const int k0 = kb * 32;
        if (threadIdx.x == 0) {
            if (kb + 1 < NK) {
                tdm_load_2d_bf16(lds_offset_of(&ldsB[(kb + 1) & 1][0]),
                                 wbase + k0 + 32, 32, 64, D_MODEL);
                __builtin_amdgcn_s_wait_tensorcnt(1);   // current buffer ready
            } else {
                __builtin_amdgcn_s_wait_tensorcnt(0);
            }
        }
        __syncthreads();                                // B panel visible to all waves

        const ushort_t* ap = arow + k0 + half * 16;
        v16bf afrag = cat8(*(const v8bf*)(ap), *(const v8bf*)(ap + 8));
        if (k0 + 256 < D_MODEL)
            __builtin_prefetch(arow + k0 + 256, 0, 3);

        const ushort_t* bl = &ldsB[kb & 1][0];
        #pragma unroll
        for (int t = 0; t < 4; ++t) {
            v16bf bfrag = *(const v16bf*)(bl + (t * 16 + l15) * 32 + half * 16);
            acc[t] = wmma_bf16(afrag, bfrag, acc[t]);
        }
        __syncthreads();                                // reads done before buffer re-targeted
    }

    if (mode == 0) {
        #pragma unroll
        for (int t = 0; t < 4; ++t)
            #pragma unroll
            for (int g = 0; g < 8; ++g) {
                int m = m0 + g + half * 8;
                int n = n0 + t * 16 + l15;
                C[(size_t)m * D_MODEL + n] = acc[t][g];
            }
    } else { // mode 2: V transposed per head
        #pragma unroll
        for (int t = 0; t < 4; ++t)
            #pragma unroll
            for (int g = 0; g < 8; ++g) {
                int m = m0 + g + half * 8;
                int n = n0 + t * 16 + l15;
                int b = m >> 11, s = m & (SEQ - 1);
                int h = n >> 6,  d = n & (HDIM - 1);
                int bh = b * NHEADS + h;
                VT[((size_t)bh * HDIM + d) * SEQ + s] = f2bfu(acc[t][g]);
            }
    }
}

// ---------------------------------------------------------------- RoPE: f32 Q/K -> bf16 per-head [B*H][S][64]
__global__ void rope_pack(const float* __restrict__ QF, const float* __restrict__ KF,
                          ushort_t* __restrict__ QB, ushort_t* __restrict__ KB) {
    int tid = blockIdx.x * blockDim.x + threadIdx.x;       // 2^21 threads
    int d = tid & 31;
    int h = (tid >> 5) & 15;
    int s = (tid >> 9) & (SEQ - 1);
    int b = tid >> 20;

    float inv_freq = __expf((float)d * -0.28782313662425573f); // 10000^(-d/32)
    float ang = (float)s * inv_freq;
    float c = __cosf(ang), sn = __sinf(ang);

    size_t row = (size_t)(b * SEQ + s) * D_MODEL + h * HDIM + d;
    size_t dst = ((size_t)(b * NHEADS + h) * SEQ + s) * HDIM + d;

    float q0 = QF[row], q1 = QF[row + 32];
    QB[dst]      = f2bfu(q0 * c - q1 * sn);
    QB[dst + 32] = f2bfu(q1 * c + q0 * sn);

    float k0 = KF[row], k1 = KF[row + 32];
    KB[dst]      = f2bfu(k0 * c - k1 * sn);
    KB[dst + 32] = f2bfu(k1 * c + k0 * sn);
}

// ---------------------------------------------------------------- Flash attention (causal, online softmax)
// K/V tiles staged by TDM (double-buffered, overlapped with WMMA); P reshaped C->A layout via LDS.
__global__ __launch_bounds__(128) void flash_attn(const ushort_t* __restrict__ QB,
                                                  const ushort_t* __restrict__ KB,
                                                  const ushort_t* __restrict__ VT,
                                                  ushort_t* __restrict__ AB) {
    __shared__ __attribute__((aligned(64))) ushort_t ldsP[4 * 16 * 32];   // per-wave P tile
    __shared__ __attribute__((aligned(64))) ushort_t ldsK[2][32 * 64];    // keys x hd
    __shared__ __attribute__((aligned(64))) ushort_t ldsV[2][64 * 32];    // hd x keys

    const int bh   = blockIdx.y;
    const int wave = threadIdx.x >> 5;
    const int lane = threadIdx.x & 31;
    const int half = lane >> 4;
    const int l15  = lane & 15;
    const int q0   = blockIdx.x * 64 + wave * 16;

    const ushort_t* kbase = KB + (size_t)bh * SEQ * HDIM;
    const ushort_t* vbase = VT + (size_t)bh * HDIM * SEQ;

    // Q fragments (reused across the whole key loop)
    v16bf aq[2];
    #pragma unroll
    for (int kk = 0; kk < 2; ++kk) {
        const ushort_t* qp = QB + ((size_t)bh * SEQ + q0 + l15) * HDIM + kk * 32 + half * 16;
        aq[kk] = cat8(*(const v8bf*)qp, *(const v8bf*)(qp + 8));
    }

    v8f o[4];
    #pragma unroll
    for (int t = 0; t < 4; ++t) o[t] = (v8f){};
    float mx[8], lsum[8];
    #pragma unroll
    for (int g = 0; g < 8; ++g) { mx[g] = -1e30f; lsum[g] = 0.0f; }

    const int nkb = blockIdx.x * 2 + 2;   // uniform per block

    if (threadIdx.x == 0) {
        tdm_load_2d_bf16(lds_offset_of(&ldsK[0][0]), kbase, 64, 32, HDIM); // 32 key rows, contiguous
        tdm_load_2d_bf16(lds_offset_of(&ldsV[0][0]), vbase, 32, 64, SEQ);  // 64 hd rows, strided
    }

    for (int kb = 0; kb < nkb; ++kb) {
        const int k0 = kb * 32;

        if (threadIdx.x == 0) {
            if (kb + 1 < nkb) {
                tdm_load_2d_bf16(lds_offset_of(&ldsK[(kb + 1) & 1][0]),
                                 kbase + (size_t)(k0 + 32) * HDIM, 64, 32, HDIM);
                tdm_load_2d_bf16(lds_offset_of(&ldsV[(kb + 1) & 1][0]),
                                 vbase + k0 + 32, 32, 64, SEQ);
                __builtin_amdgcn_s_wait_tensorcnt(2);   // current K/V pair done, next in flight
            } else {
                __builtin_amdgcn_s_wait_tensorcnt(0);
            }
        }
        __syncthreads();                                // tiles visible to all waves

        const ushort_t* kl = &ldsK[kb & 1][0];
        const ushort_t* vl = &ldsV[kb & 1][0];

        v8f s0 = (v8f){}, s1 = (v8f){};
        #pragma unroll
        for (int kk = 0; kk < 2; ++kk) {
            v16bf b0 = *(const v16bf*)(kl + (l15)      * HDIM + kk * 32 + half * 16);
            v16bf b1 = *(const v16bf*)(kl + (16 + l15) * HDIM + kk * 32 + half * 16);
            s0 = wmma_bf16(aq[kk], b0, s0);
            s1 = wmma_bf16(aq[kk], b1, s1);
        }

        #pragma unroll
        for (int g = 0; g < 8; ++g) {
            int qi = q0 + g + half * 8;
            int j0 = k0 + l15, j1 = j0 + 16;
            float v0 = s0[g] * 0.125f;          // 1/sqrt(64)
            float v1 = s1[g] * 0.125f;
            if (j0 > qi) v0 = -1e30f;
            if (j1 > qi) v1 = -1e30f;

            float t = fmaxf(v0, v1);
            t = fmaxf(t, __shfl_xor(t, 1));
            t = fmaxf(t, __shfl_xor(t, 2));
            t = fmaxf(t, __shfl_xor(t, 4));
            t = fmaxf(t, __shfl_xor(t, 8));
            float mn = fmaxf(mx[g], t);
            float corr = __expf(mx[g] - mn);
            mx[g] = mn;

            float p0 = __expf(v0 - mn);
            float p1 = __expf(v1 - mn);
            float rs = p0 + p1;
            rs += __shfl_xor(rs, 1);
            rs += __shfl_xor(rs, 2);
            rs += __shfl_xor(rs, 4);
            rs += __shfl_xor(rs, 8);
            lsum[g] = lsum[g] * corr + rs;

            #pragma unroll
            for (int t4 = 0; t4 < 4; ++t4) o[t4][g] = o[t4][g] * corr;

            int row = g + half * 8;
            ldsP[wave * 512 + row * 32 + l15]      = f2bfu(p0);
            ldsP[wave * 512 + row * 32 + 16 + l15] = f2bfu(p1);
        }
        // per-wave private P region: in-wave DS ordering suffices, no block barrier needed
        const ushort_t* pp = &ldsP[wave * 512 + l15 * 32 + half * 16];
        v16bf ap = cat8(*(const v8bf*)pp, *(const v8bf*)(pp + 8));

        #pragma unroll
        for (int t = 0; t < 4; ++t) {
            int vn = t * 16 + l15;   // hd column
            v16bf bv = *(const v16bf*)(vl + vn * 32 + half * 16);
            o[t] = wmma_bf16(ap, bv, o[t]);
        }
        __syncthreads();                                // K/V reads done before buffer re-targeted
    }

    const int b = bh >> 4, h = bh & 15;
    #pragma unroll
    for (int g = 0; g < 8; ++g) {
        float inv = 1.0f / lsum[g];
        int seq = q0 + g + half * 8;
        #pragma unroll
        for (int t = 0; t < 4; ++t) {
            int n = t * 16 + l15;
            AB[((size_t)(b * SEQ + seq)) * D_MODEL + h * HDIM + n] = f2bfu(o[t][g] * inv);
        }
    }
}

// ---------------------------------------------------------------- gate: out = o * sigmoid(g + bg)
__global__ void gate_final(const float* __restrict__ OF, const float* __restrict__ GF,
                           const float* __restrict__ bg, float* __restrict__ out, int n) {
    for (int i = blockIdx.x * blockDim.x + threadIdx.x; i < n; i += gridDim.x * blockDim.x) {
        float gp = GF[i] + bg[i & (D_MODEL - 1)];
        float sg = 1.0f / (1.0f + __expf(-gp));
        out[i] = OF[i] * sg;
    }
}

// ---------------------------------------------------------------- launch
extern "C" void kernel_launch(void* const* d_in, const int* in_sizes, int n_in,
                              void* d_out, int out_size, void* d_ws, size_t ws_size,
                              hipStream_t stream) {
    const float* x  = (const float*)d_in[0];
    const float* bg = (const float*)d_in[6];

    char* ws = (char*)d_ws;
    const size_t Mi = 1ull << 20;
    ushort_t* XB = (ushort_t*)(ws + 0);          //  8 MiB bf16 x
    ushort_t* WB = (ushort_t*)(ws + 8  * Mi);    // 10 MiB bf16 weights (5 x 1 M elem)
    float*    QF = (float*)   (ws + 18 * Mi);    // 16 MiB
    float*    KF = (float*)   (ws + 34 * Mi);    // 16 MiB
    ushort_t* VT = (ushort_t*)(ws + 50 * Mi);    //  8 MiB (per-head transposed V)
    ushort_t* QB = (ushort_t*)(ws + 58 * Mi);    //  8 MiB
    ushort_t* KB = (ushort_t*)(ws + 66 * Mi);    //  8 MiB
    ushort_t* AB = (ushort_t*)(ws + 74 * Mi);    //  8 MiB (attn out bf16)  -> peak 82 MiB
    float*    OF = (float*)   (ws + 18 * Mi);    // reuse QF (free after rope)
    ushort_t* OB = (ushort_t*)(ws + 34 * Mi);    // reuse KF
    float*    GF = (float*)   (ws + 42 * Mi);    // reuse KF tail + VT (free after flash)

    convert_bf16<<<4096, 256, 0, stream>>>(x, XB, NELEM);
    for (int w = 0; w < 5; ++w)
        convert_bf16<<<1024, 256, 0, stream>>>((const float*)d_in[1 + w],
                                               WB + (size_t)w * D_MODEL * D_MODEL,
                                               D_MODEL * D_MODEL);

    dim3 ggrid(D_MODEL / 64, MTOT / 64);  // 16 x 64
    gemm_bf16_wmma<<<ggrid, 128, 0, stream>>>(XB, WB + 0 * 1048576, QF, nullptr, 0);
    gemm_bf16_wmma<<<ggrid, 128, 0, stream>>>(XB, WB + 1 * 1048576, KF, nullptr, 0);
    gemm_bf16_wmma<<<ggrid, 128, 0, stream>>>(XB, WB + 2 * 1048576, nullptr, VT, 2);

    rope_pack<<<8192, 256, 0, stream>>>(QF, KF, QB, KB);

    dim3 fgrid(SEQ / 64, BATCH * NHEADS);  // 32 x 32
    flash_attn<<<fgrid, 128, 0, stream>>>(QB, KB, VT, AB);

    gemm_bf16_wmma<<<ggrid, 128, 0, stream>>>(AB, WB + 3 * 1048576, OF, nullptr, 0);
    convert_bf16<<<4096, 256, 0, stream>>>(OF, OB, NELEM);
    gemm_bf16_wmma<<<ggrid, 128, 0, stream>>>(OB, WB + 4 * 1048576, GF, nullptr, 0);

    gate_final<<<4096, 256, 0, stream>>>(OF, GF, bg, (float*)d_out, NELEM);
}
